// GroupedQueryAttention_58892591562761
// MI455X (gfx1250) — compile-verified, matrix operand order checked by reading, not run on hardware
//
#include <hip/hip_runtime.h>
#include <math.h>

typedef __bf16 bf16_t;
typedef __attribute__((ext_vector_type(16))) __bf16 v16bf;
typedef __attribute__((ext_vector_type(8)))  __bf16 v8bf;
typedef __attribute__((ext_vector_type(8)))  float  v8f;

namespace {
constexpr int kB   = 2;
constexpr int kS   = 2048;
constexpr int kDM  = 1024;
constexpr int kNH  = 16;
constexpr int kNKV = 4;
constexpr int kGRP = 4;            // kNH / kNKV
constexpr int kDH  = 64;
constexpr int kM   = kB * kS;      // 4096 rows
constexpr int kNQKV = kNH*kDH + 2*kNKV*kDH;  // 1536
constexpr float kScale = 0.125f;   // 1/sqrt(64)
constexpr float kCap   = 15.0f;
constexpr int kWin   = 256;
constexpr int kNGlob = 4;
}

// ---- helpers ---------------------------------------------------------------

__device__ __forceinline__ bf16_t f2bf(float f) {
  unsigned u = __float_as_uint(f);
  unsigned short h = (unsigned short)((u + 0x7FFFu + ((u >> 16) & 1u)) >> 16);
  union { unsigned short s; bf16_t b; } cv;
  cv.s = h;
  return cv.b;
}

// Build one 16x32 bf16 WMMA fragment for this lane (A or B operand).
// ISA layout: lane-half (lane>=16) adds +8 to k; elems 0..7 -> k=kbase..+7,
// elems 8..15 -> k=kbase+16..+23.  kbase must already include half*8.
__device__ __forceinline__ v16bf load_frag(const bf16_t* __restrict__ row, int kbase) {
  v8bf lo = *reinterpret_cast<const v8bf*>(row + kbase);
  v8bf hi = *reinterpret_cast<const v8bf*>(row + kbase + 16);
  v16bf f;
#pragma unroll
  for (int i = 0; i < 8; ++i) { f[i] = lo[i]; f[i + 8] = hi[i]; }
  return f;
}

__device__ __forceinline__ v8f wmma_bf16(v16bf a, v16bf b, v8f c) {
  return __builtin_amdgcn_wmma_f32_16x16x32_bf16(false, a, false, b, (short)0, c,
                                                 false, false);
}

__device__ __forceinline__ float redmax16(float v) {
#pragma unroll
  for (int m = 1; m <= 8; m <<= 1) v = fmaxf(v, __shfl_xor(v, m, 32));
  return v;
}
__device__ __forceinline__ float redsum16(float v) {
#pragma unroll
  for (int m = 1; m <= 8; m <<= 1) v += __shfl_xor(v, m, 32);
  return v;
}
__device__ __forceinline__ float redsum32(float v) {
#pragma unroll
  for (int m = 1; m <= 16; m <<= 1) v += __shfl_xor(v, m, 32);
  return v;
}

// ---- conversion / packing kernels ------------------------------------------

__global__ void cvt_x_kernel(const float* __restrict__ x, bf16_t* __restrict__ xb, int n) {
  int i = blockIdx.x * blockDim.x + threadIdx.x;
  if (i < n) xb[i] = f2bf(x[i]);
}

// Pack [Wq|Wk|Wv] transposed: Wp[n][k], n in [0,1536), k in [0,1024)
__global__ void pack_qkv_w_kernel(const float* __restrict__ Wq,
                                  const float* __restrict__ Wk,
                                  const float* __restrict__ Wv,
                                  bf16_t* __restrict__ Wp) {
  int i = blockIdx.x * blockDim.x + threadIdx.x;
  if (i >= kNQKV * kDM) return;
  int n = i >> 10, k = i & 1023;
  float v;
  if (n < kNH * kDH)                   v = Wq[(size_t)k * (kNH * kDH) + n];
  else if (n < kNH * kDH + kNKV * kDH) v = Wk[(size_t)k * (kNKV * kDH) + (n - kNH * kDH)];
  else                                 v = Wv[(size_t)k * (kNKV * kDH) + (n - kNH * kDH - kNKV * kDH)];
  Wp[i] = f2bf(v);
}

__global__ void pack_wo_kernel(const float* __restrict__ Wo, bf16_t* __restrict__ Wp) {
  int i = blockIdx.x * blockDim.x + threadIdx.x;
  if (i >= kDM * kDM) return;
  int n = i >> 10, k = i & 1023;
  Wp[i] = f2bf(Wo[(size_t)k * kDM + n]);
}

// ---- bf16 WMMA GEMM: C[M,N] f32 = A[M,K] * Bp[N,K]^T -----------------------
// one wave computes a 32(M) x 64(N) tile (2x4 WMMA sub-tiles) so each B
// fragment feeds two WMMAs: 12 b128 loads per 8 WMMAs per 32-k step.
// blockDim = 256 (8 waves)

__global__ __launch_bounds__(256)
void wmma_gemm_kernel(const bf16_t* __restrict__ A, const bf16_t* __restrict__ Bp,
                      float* __restrict__ C, int M, int N, int K) {
  int wave = (int)((blockIdx.x * blockDim.x + threadIdx.x) >> 5);
  int lane = threadIdx.x & 31;
  int hl = lane >> 4;        // half of wave
  int r  = lane & 15;
  int ngrp = N >> 6;         // groups of 64 columns
  int mtile = wave / ngrp;
  int ncol  = (wave % ngrp) * 64;
  if (mtile * 32 >= M) return;
  int m0 = mtile * 32;

  const bf16_t* arow0 = A + (size_t)(m0 + r) * K;
  const bf16_t* arow1 = A + (size_t)(m0 + 16 + r) * K;
  v8f acc[2][4] = {};

  for (int k0 = 0; k0 < K; k0 += 32) {
    // prefetch next k block of A rows into cache (global_prefetch_b8)
    if (k0 + 32 < K) {
      __builtin_prefetch(arow0 + k0 + 32 + hl * 8, 0, 3);
      __builtin_prefetch(arow1 + k0 + 32 + hl * 8, 0, 3);
    }
    v16bf a0 = load_frag(arow0, k0 + hl * 8);
    v16bf a1 = load_frag(arow1, k0 + hl * 8);
#pragma unroll
    for (int j = 0; j < 4; ++j) {
      const bf16_t* brow = Bp + (size_t)(ncol + j * 16 + r) * K;
      v16bf bfrag = load_frag(brow, k0 + hl * 8);
      acc[0][j] = wmma_bf16(a0, bfrag, acc[0][j]);
      acc[1][j] = wmma_bf16(a1, bfrag, acc[1][j]);
    }
  }

#pragma unroll
  for (int t = 0; t < 2; ++t)
#pragma unroll
    for (int j = 0; j < 4; ++j)
#pragma unroll
      for (int i = 0; i < 8; ++i)
        C[(size_t)(m0 + t * 16 + i + 8 * hl) * N + ncol + j * 16 + r] = acc[t][j][i];
}

// ---- L2-normalize + RoPE + layout kernel -----------------------------------
// qkv: f32 [kM, 1536].  Outputs:
//   Qb [B][NH][S][64] bf16, Kb [B][NKV][S][64] bf16, Vt [B][NKV][64][S] bf16.

__global__ __launch_bounds__(256)
void norm_rope_kernel(const float* __restrict__ qkv,
                      const float* __restrict__ cosb, const float* __restrict__ sinb,
                      bf16_t* __restrict__ Qb, bf16_t* __restrict__ Kb,
                      bf16_t* __restrict__ Vt) {
  int gw = (int)((blockIdx.x * blockDim.x + threadIdx.x) >> 5);
  int lane = threadIdx.x & 31;
  const int QW = kB * kNH * kS;
  const int KW = kB * kNKV * kS;

  if (gw < QW) {
    int s = gw % kS, h = (gw / kS) % kNH, b = gw / (kS * kNH);
    const float* src = qkv + (size_t)(b * kS + s) * kNQKV + h * kDH;
    float x1 = src[lane], x2 = src[lane + 32];
    float inv = 1.0f / (sqrtf(redsum32(x1 * x1 + x2 * x2)) + 1e-8f);
    x1 *= inv; x2 *= inv;
    float c = cosb[s * 32 + lane], sn = sinb[s * 32 + lane];
    bf16_t* dst = Qb + ((size_t)((b * kNH + h) * kS) + s) * kDH;
    dst[lane]      = f2bf(x1 * c - x2 * sn);
    dst[lane + 32] = f2bf(x1 * sn + x2 * c);
  } else if (gw < QW + KW) {
    int t = gw - QW;
    int s = t % kS, kh = (t / kS) % kNKV, b = t / (kS * kNKV);
    const float* src = qkv + (size_t)(b * kS + s) * kNQKV + kNH * kDH + kh * kDH;
    float x1 = src[lane], x2 = src[lane + 32];
    float inv = 1.0f / (sqrtf(redsum32(x1 * x1 + x2 * x2)) + 1e-8f);
    x1 *= inv; x2 *= inv;
    float c = cosb[s * 32 + lane], sn = sinb[s * 32 + lane];
    bf16_t* dst = Kb + ((size_t)((b * kNKV + kh) * kS) + s) * kDH;
    dst[lane]      = f2bf(x1 * c - x2 * sn);
    dst[lane + 32] = f2bf(x1 * sn + x2 * c);
  } else if (gw < QW + 2 * KW) {
    int t = gw - QW - KW;
    int s = t % kS, kh = (t / kS) % kNKV, b = t / (kS * kNKV);
    const float* src = qkv + (size_t)(b * kS + s) * kNQKV + (kNH + kNKV) * kDH + kh * kDH;
    bf16_t* dst = Vt + (size_t)(b * kNKV + kh) * kDH * kS;
    dst[(size_t)lane * kS + s]        = f2bf(src[lane]);
    dst[(size_t)(lane + 32) * kS + s] = f2bf(src[lane + 32]);
  }
}

// ---- sliding-window flash attention with softcap ---------------------------
// one wave (one block of 32) per (b, h, 16-row q tile)

__device__ __forceinline__ bool att_valid(int q, int k) {
  return ((k <= q) && (k > q - kWin)) || (k < kNGlob);
}

__global__ __launch_bounds__(32)
void attn_kernel(const bf16_t* __restrict__ Qb, const bf16_t* __restrict__ Kb,
                 const bf16_t* __restrict__ Vt, bf16_t* __restrict__ ctx) {
  __shared__ float Plds[16][33];

  const int qtiles = kS / 16;
  int wid = blockIdx.x;
  int qt = wid % qtiles;
  int h  = (wid / qtiles) % kNH;
  int b  = wid / (qtiles * kNH);
  int kh = h / kGRP;
  int lane = threadIdx.x & 31;
  int hl = lane >> 4, r = lane & 15;
  int q0 = qt * 16;

  const bf16_t* Qrow  = Qb + ((size_t)((b * kNH + h) * kS) + q0 + r) * kDH;
  const bf16_t* Kbase = Kb + (size_t)(b * kNKV + kh) * kS * kDH;
  const bf16_t* Vbase = Vt + (size_t)(b * kNKV + kh) * kDH * kS;

  float mrow[8], lrow[8];
  v8f acc[4] = {};
#pragma unroll
  for (int i = 0; i < 8; ++i) { mrow[i] = -3.0e38f; lrow[i] = 0.0f; }

  int kstart = q0 - (kWin - 1);
  if (kstart < 0) kstart = 0;
  kstart &= ~31;
  bool global_phase = (kstart > 0);
  int kb = global_phase ? 0 : kstart;

  for (;;) {
    // --- scores for keys [kb, kb+31] ---
    v8f sA = {}, sB = {};
    const bf16_t* KrA = Kbase + (size_t)(kb + r) * kDH;
    const bf16_t* KrB = Kbase + (size_t)(kb + 16 + r) * kDH;
#pragma unroll
    for (int kk = 0; kk < kDH; kk += 32) {
      v16bf qf  = load_frag(Qrow, kk + hl * 8);
      v16bf kfA = load_frag(KrA,  kk + hl * 8);
      v16bf kfB = load_frag(KrB,  kk + hl * 8);
      sA = wmma_bf16(qf, kfA, sA);
      sB = wmma_bf16(qf, kfB, sB);
    }

    // --- softcap + mask + online softmax (8 rows per lane-half) ---
#pragma unroll
    for (int i = 0; i < 8; ++i) {
      int q = q0 + i + 8 * hl;
      int ka = kb + r, kc = kb + 16 + r;
      float a = kCap * tanhf(sA[i] * (kScale / kCap));
      float c = kCap * tanhf(sB[i] * (kScale / kCap));
      if (!att_valid(q, ka)) a = -1.0e9f;
      if (!att_valid(q, kc)) c = -1.0e9f;
      float mx   = redmax16(fmaxf(a, c));
      float mnew = fmaxf(mrow[i], mx);
      float sc   = __expf(mrow[i] - mnew);
      float pa   = __expf(a - mnew);
      float pc   = __expf(c - mnew);
      lrow[i] = lrow[i] * sc + redsum16(pa + pc);
      mrow[i] = mnew;
#pragma unroll
      for (int j = 0; j < 4; ++j) acc[j][i] *= sc;
      Plds[i + 8 * hl][r]      = pa;
      Plds[i + 8 * hl][16 + r] = pc;
    }
    __syncthreads();

    // --- C-layout P -> A-fragment (16x32 bf16) via LDS ---
    v16bf pf;
#pragma unroll
    for (int e = 0; e < 8; ++e) {
      pf[e]     = f2bf(Plds[r][hl * 8 + e]);
      pf[e + 8] = f2bf(Plds[r][16 + hl * 8 + e]);
    }
    __syncthreads();

    // --- context accumulate: acc += P * V ---
#pragma unroll
    for (int j = 0; j < 4; ++j) {
      const bf16_t* vrow = Vbase + (size_t)(j * 16 + r) * kS + kb;
      v16bf vf = load_frag(vrow, hl * 8);
      acc[j] = wmma_bf16(pf, vf, acc[j]);
    }

    if (global_phase) { kb = kstart; global_phase = false; }
    else { kb += 32; if (kb >= q0 + 16) break; }
  }

  // --- normalize and write bf16 context in [b, s, h*64+d] layout ---
#pragma unroll
  for (int j = 0; j < 4; ++j)
#pragma unroll
    for (int i = 0; i < 8; ++i) {
      int q = q0 + i + 8 * hl;
      float o = acc[j][i] / lrow[i];
      ctx[(size_t)(b * kS + q) * (kNH * kDH) + h * kDH + j * 16 + r] = f2bf(o);
    }
}

// ---- host launch ------------------------------------------------------------

extern "C" void kernel_launch(void* const* d_in, const int* in_sizes, int n_in,
                              void* d_out, int out_size, void* d_ws, size_t ws_size,
                              hipStream_t stream) {
  (void)in_sizes; (void)n_in; (void)out_size; (void)ws_size;
  const float* x    = (const float*)d_in[0];
  const float* cosb = (const float*)d_in[1];
  const float* sinb = (const float*)d_in[2];
  // d_in[3] = mask (recomputed analytically on device)
  const float* Wq = (const float*)d_in[4];
  const float* Wk = (const float*)d_in[5];
  const float* Wv = (const float*)d_in[6];
  const float* Wo = (const float*)d_in[7];
  float* out = (float*)d_out;

  // workspace partition (all 256B aligned by construction)
  char* p = (char*)d_ws;
  bf16_t* xb   = (bf16_t*)p;              p += (size_t)kM * kDM * sizeof(bf16_t);
  bf16_t* wp   = (bf16_t*)p;              p += (size_t)kNQKV * kDM * sizeof(bf16_t);
  bf16_t* wop  = (bf16_t*)p;              p += (size_t)kDM * kDM * sizeof(bf16_t);
  float*  qkv  = (float*)p;               p += (size_t)kM * kNQKV * sizeof(float);
  bf16_t* Qb   = (bf16_t*)p;              p += (size_t)kB * kNH  * kS * kDH * sizeof(bf16_t);
  bf16_t* Kb   = (bf16_t*)p;              p += (size_t)kB * kNKV * kS * kDH * sizeof(bf16_t);
  bf16_t* Vt   = (bf16_t*)p;              p += (size_t)kB * kNKV * kDH * kS * sizeof(bf16_t);
  bf16_t* ctx  = (bf16_t*)p;              p += (size_t)kM * kDM * sizeof(bf16_t);

  // 1. conversions / packing
  cvt_x_kernel<<<(kM * kDM + 255) / 256, 256, 0, stream>>>(x, xb, kM * kDM);
  pack_qkv_w_kernel<<<(kNQKV * kDM + 255) / 256, 256, 0, stream>>>(Wq, Wk, Wv, wp);
  pack_wo_kernel<<<(kDM * kDM + 255) / 256, 256, 0, stream>>>(Wo, wop);

  // 2. fused QKV projection: [4096,1024] x [1024,1536]^T -> f32 [4096,1536]
  {
    int waves = (kM / 32) * (kNQKV / 64);   // 3072
    wmma_gemm_kernel<<<waves / 8, 256, 0, stream>>>(xb, wp, qkv, kM, kNQKV, kDM);
  }

  // 3. l2-norm + rope + layout
  {
    int waves = kB * kNH * kS + 2 * kB * kNKV * kS;
    norm_rope_kernel<<<(waves + 7) / 8, 256, 0, stream>>>(qkv, cosb, sinb, Qb, Kb, Vt);
  }

  // 4. windowed flash attention
  attn_kernel<<<kB * kNH * (kS / 16), 32, 0, stream>>>(Qb, Kb, Vt, ctx);

  // 5. output projection: [4096,1024] x [1024,1024]^T -> f32 out
  {
    int waves = (kM / 32) * (kDM / 64);     // 2048
    wmma_gemm_kernel<<<waves / 8, 256, 0, stream>>>(ctx, wop, out, kM, kDM, kDM);
  }
}
